// GQA_RoPE_Attention_10797547782825
// MI455X (gfx1250) — matrix operand order repaired and checked by reading, not compiled, not run on hardware
//
#include <hip/hip_runtime.h>
#include <hip/hip_bf16.h>

typedef __attribute__((ext_vector_type(16))) __bf16 bf16x16;
typedef __attribute__((ext_vector_type(8)))  __bf16 bf16x8;
typedef __attribute__((ext_vector_type(8)))  float  f32x8;
typedef __attribute__((ext_vector_type(4)))  unsigned int v4u;
typedef __attribute__((ext_vector_type(8)))  int v8i;
typedef __attribute__((ext_vector_type(4)))  int v4i;

#define B_   2
#define L_   2048
#define D_   1024
#define H_   16
#define G_   4
#define HD_  64
#define WIN_ 256

// LDS row strides (elements) chosen for conflict-free ds_load_b128 fragment reads
#define KST  72   // 32 keys x 64 hd, 144B rows: banks 36*r mod 64 distinct
#define VST  40   // 64 hd x 32 keys, 80B rows: banks 20*r mod 64 distinct
#define PST  40
#define KSZ  (32 * KST)   // elements per K stage buffer
#define VSZ  (64 * VST)   // elements per V stage buffer

// Workspace byte offsets
#define WQT_OFF  (size_t)0
#define WKT_OFF  (size_t)2097152
#define WVT_OFF  (size_t)2621440
#define WOT_OFF  (size_t)3145728
#define QBF_OFF  (size_t)5242880
#define KBF_OFF  (size_t)13631488
#define VTB_OFF  (size_t)15728640
#define ATTN_OFF (size_t)17825792

static __device__ __forceinline__ bf16x16 mk16(bf16x8 lo, bf16x8 hi) {
    bf16x16 r;
#pragma unroll
    for (int i = 0; i < 8; ++i) { r[i] = lo[i]; r[i + 8] = hi[i]; }
    return r;
}

static __device__ __forceinline__ f32x8 wmma_bf16(bf16x16 a, bf16x16 b, f32x8 c) {
    return __builtin_amdgcn_wmma_f32_16x16x32_bf16(false, a, false, b, (short)0, c, false, false);
}

static __device__ __forceinline__ bf16x16 cvt16(const float* p) {
    float4 f0 = *(const float4*)(p);
    float4 f1 = *(const float4*)(p + 4);
    float4 f2 = *(const float4*)(p + 8);
    float4 f3 = *(const float4*)(p + 12);
    bf16x16 a;
    a[0]=(__bf16)f0.x; a[1]=(__bf16)f0.y; a[2]=(__bf16)f0.z; a[3]=(__bf16)f0.w;
    a[4]=(__bf16)f1.x; a[5]=(__bf16)f1.y; a[6]=(__bf16)f1.z; a[7]=(__bf16)f1.w;
    a[8]=(__bf16)f2.x; a[9]=(__bf16)f2.y; a[10]=(__bf16)f2.z; a[11]=(__bf16)f2.w;
    a[12]=(__bf16)f3.x; a[13]=(__bf16)f3.y; a[14]=(__bf16)f3.z; a[15]=(__bf16)f3.w;
    return a;
}

// ---- Tensor Data Mover: 2-D tile (data_size = 4B) with LDS pad -------------
#if __has_builtin(__builtin_amdgcn_tensor_load_to_lds)
#define HAVE_TDM 1
static __device__ __forceinline__ void tdm_load_2d(
    unsigned lds_off, const void* gaddr,
    unsigned dim0_dw, unsigned rows, unsigned stride_dw,
    unsigned pad_interval_code, unsigned pad_amount_code)
{
    unsigned long long ga = (unsigned long long)gaddr;
    v4u g0;
    g0[0] = 1u;                                   // count=1, user descriptor
    g0[1] = lds_off;                              // LDS byte address
    g0[2] = (unsigned)ga;                         // global_addr[31:0]
    g0[3] = (unsigned)(ga >> 32) | 0x80000000u;   // global_addr[56:32] | type=2
    v8i g1;
    g1[0] = (int)((2u << 16) | (1u << 20) |       // data_size=4B, pad_enable
                  (pad_interval_code << 22) | (pad_amount_code << 25));
    g1[1] = (int)((dim0_dw & 0xFFFFu) << 16);     // tensor_dim0 lo16 (barrier addr=0)
    g1[2] = (int)((dim0_dw >> 16) | ((rows & 0xFFFFu) << 16));  // td0 hi | td1 lo
    g1[3] = (int)((rows >> 16) | (dim0_dw << 16));              // td1 hi | tile_dim0
    g1[4] = (int)(rows & 0xFFFFu);                // tile_dim1 (tile_dim2=0)
    g1[5] = (int)stride_dw;                       // tensor_dim0_stride lo32
    g1[6] = 0;
    g1[7] = 0;
    v4i z4 = {};
    v8i z8 = {};
    __builtin_amdgcn_tensor_load_to_lds(g0, g1, z4, z4, z8, 0);   // clang-23 6-arg form
}
#endif

static __device__ __forceinline__ void wait_tensorcnt0() {
#if __has_builtin(__builtin_amdgcn_s_wait_tensorcnt)
    __builtin_amdgcn_s_wait_tensorcnt((short)0);
#else
    asm volatile("s_wait_tensorcnt 0x0" ::: "memory");
#endif
}
static __device__ __forceinline__ void wait_tensorcnt2() {
#if __has_builtin(__builtin_amdgcn_s_wait_tensorcnt)
    __builtin_amdgcn_s_wait_tensorcnt((short)2);
#else
    asm volatile("s_wait_tensorcnt 0x2" ::: "memory");
#endif
}

// ---------------------------------------------------------------------------
// Kernel 0: transpose + convert f32 weight (K=1024 rows x N cols) -> bf16 WT[n][k]
// ---------------------------------------------------------------------------
__global__ void wconv_k(const float* __restrict__ W, __bf16* __restrict__ WT, int N) {
    int idx = blockIdx.x * 256 + threadIdx.x;
    int n = idx >> 10;
    int k = idx & 1023;
    WT[(size_t)n * 1024 + k] = (__bf16)W[(size_t)k * N + n];
}

// ---------------------------------------------------------------------------
// Kernel 1: fused QKV projection + bias + RoPE.
// One wave computes a 32(M) x 64(N) strip: 2 A-frags x 4 B-frags = 8 WMMA/chunk.
// Fused N: [Q 0..1023 | K 1024..1279 | V 1280..1535] -> 24 strips of 64.
// ---------------------------------------------------------------------------
__global__ __launch_bounds__(32) void qkv_rope_k(
    const float* __restrict__ x,
    const __bf16* __restrict__ WqT, const __bf16* __restrict__ WkT, const __bf16* __restrict__ WvT,
    const float* __restrict__ bq, const float* __restrict__ bk, const float* __restrict__ bv,
    __bf16* __restrict__ qbf, __bf16* __restrict__ kbf, __bf16* __restrict__ vT)
{
    const int lane = threadIdx.x;
    const int lr   = lane & 15;
    const int half = lane >> 4;

    const int ns = blockIdx.x;          // 0..23 strips of 64 columns
    const int m0 = blockIdx.y * 32;     // 0..127 strips of 32 rows
    const int b  = m0 >> 11;
    const int l0 = m0 & (L_ - 1);

    const __bf16* WT;
    const float*  bias;
    int n0, region;
    if (ns < 16)      { region = 0; n0 = ns * 64;        WT = WqT; bias = bq; }
    else if (ns < 20) { region = 1; n0 = (ns - 16) * 64; WT = WkT; bias = bk; }
    else              { region = 2; n0 = (ns - 20) * 64; WT = WvT; bias = bv; }
    const int hg = n0 >> 6;             // head (Q) or group (K/V); strip == one head

    f32x8 acc[2][4] = {};
    const float* arow0 = x + (size_t)(m0 + lr) * D_ + half * 16;
    const float* arow1 = x + (size_t)(m0 + 16 + lr) * D_ + half * 16;
    const __bf16* bcol = WT + (size_t)(n0 + lr) * 1024 + half * 16;

    for (int kc = 0; kc < 1024; kc += 32) {
        bf16x16 a0 = cvt16(arow0 + kc);
        bf16x16 a1 = cvt16(arow1 + kc);
#pragma unroll
        for (int nt = 0; nt < 4; ++nt) {
            bf16x16 bm = *(const bf16x16*)(bcol + (size_t)nt * 16 * 1024 + kc);
            acc[0][nt] = wmma_bf16(a0, bm, acc[0][nt]);
            acc[1][nt] = wmma_bf16(a1, bm, acc[1][nt]);
        }
    }

#pragma unroll
    for (int nt = 0; nt < 4; ++nt) {
        const int   hd   = nt * 16 + lr;          // dim within head (strip-aligned)
        const float bvl  = bias[n0 + hd];
        const int   fidx = hd & 31;
        const float invf = __expf(-(float)fidx * 0.28782313663f);  // ln(1e4)*2/64
        const bool  even = (hd & 1) == 0;
#pragma unroll
        for (int mt2 = 0; mt2 < 2; ++mt2) {
            const int l0m = l0 + mt2 * 16;
            float c[8];
#pragma unroll
            for (int r = 0; r < 8; ++r) c[r] = acc[mt2][nt][r] + bvl;

            if (region <= 1) {  // RoPE on Q and K
#pragma unroll
                for (int r = 0; r < 8; ++r) {
                    int   mrow  = l0m + half * 8 + r;
                    float th    = (float)mrow * invf;
                    float sn    = __sinf(th), cs = __cosf(th);
                    float other = __shfl_xor(c[r], 1, 32);
                    c[r] = even ? (c[r] * cs - other * sn) : (c[r] * cs + other * sn);
                }
            }

            if (region == 0) {
                size_t base = ((size_t)(b * H_ + hg)) * L_ * HD_;
#pragma unroll
                for (int r = 0; r < 8; ++r)
                    qbf[base + (size_t)(l0m + half * 8 + r) * HD_ + hd] = (__bf16)c[r];
            } else if (region == 1) {
                size_t base = ((size_t)(b * G_ + hg)) * L_ * HD_;
#pragma unroll
                for (int r = 0; r < 8; ++r)
                    kbf[base + (size_t)(l0m + half * 8 + r) * HD_ + hd] = (__bf16)c[r];
            } else {
                bf16x8 t;
#pragma unroll
                for (int r = 0; r < 8; ++r) t[r] = (__bf16)c[r];
                *(bf16x8*)(vT + (((size_t)(b * G_ + hg)) * HD_ + hd) * L_ + l0m + half * 8) = t;
            }
        }
    }
}

// ---------------------------------------------------------------------------
// Kernel 2: banded + global attention. Block = 4 waves = 4 heads of one KV
// group over one 16-query tile. K/V 32-key stages double-buffered in LDS,
// staged by the Tensor Data Mover (TENSORcnt pipeline).
// ---------------------------------------------------------------------------
__global__ __launch_bounds__(128) void attn_k(
    const __bf16* __restrict__ qbf, const __bf16* __restrict__ kbf,
    const __bf16* __restrict__ vT,  const int* __restrict__ gp,
    float* __restrict__ attn)
{
    __shared__ __bf16 Ks[2 * KSZ];         // [stage][key][hd], padded rows
    __shared__ __bf16 Vs[2 * VSZ];         // [stage][hd][key], padded rows
    __shared__ __bf16 Pl[4 * 16 * PST];    // per-wave P tile (C->A re-layout)

    const int tid  = threadIdx.x;
    const int lane = tid & 31;
    const int wid  = tid >> 5;
    const int lr   = lane & 15;
    const int half = (lane >> 4) & 1;

    const int qt = blockIdx.x;
    const int g  = blockIdx.y;
    const int b  = blockIdx.z;
    const int h  = g * (H_ / G_) + wid;
    const int l0 = qt * 16;
    const int NKB = L_ / 32;

    const int4 gpv = *(const int4*)gp;

    const __bf16* qrow = qbf + (((size_t)(b * H_ + h)) * L_ + (l0 + lr)) * HD_ + half * 16;
    bf16x16 aq0 = mk16(*(const bf16x8*)(qrow),      *(const bf16x8*)(qrow + 8));
    bf16x16 aq1 = mk16(*(const bf16x8*)(qrow + 32), *(const bf16x8*)(qrow + 40));

    float mrun[8], lrun[8];
    f32x8 o[4] = {};
    bool  rg[8];
#pragma unroll
    for (int r = 0; r < 8; ++r) {
        mrun[r] = -1e30f; lrun[r] = 0.f;
        int mrow = l0 + half * 8 + r;
        rg[r] = (mrow == gpv.x) | (mrow == gpv.y) | (mrow == gpv.z) | (mrow == gpv.w);
    }
    const bool qglob = ((gpv.x >> 4) == qt) | ((gpv.y >> 4) == qt) |
                       ((gpv.z >> 4) == qt) | ((gpv.w >> 4) == qt);

    auto needed = [&](int kb_) -> bool {
        int s0_ = kb_ * 32;
        bool band = (s0_ <= l0 + 15 + WIN_) && (s0_ + 31 >= l0 - WIN_);
        bool kg = ((gpv.x >> 5) == kb_) | ((gpv.y >> 5) == kb_) |
                  ((gpv.z >> 5) == kb_) | ((gpv.w >> 5) == kb_);
        return band | kg | qglob;
    };

    const __bf16* kbase = kbf + ((size_t)(b * G_ + g)) * L_ * HD_;
    const __bf16* vbase = vT  + ((size_t)(b * G_ + g)) * HD_ * L_;
    __bf16* Pw = Pl + wid * 16 * PST;
#ifdef HAVE_TDM
    const unsigned ldsK = (unsigned)(unsigned long long)(void*)Ks;
    const unsigned ldsV = (unsigned)(unsigned long long)(void*)Vs;
#endif

    int kb = 0;
    while (kb < NKB && !needed(kb)) ++kb;    // first stage (diagonal guarantees one)
    int buf = 0;

#ifdef HAVE_TDM
    if (wid == 0 && kb < NKB) {
        // K tile: 32 rows x 128B contiguous; LDS rows padded to 144B
        tdm_load_2d(ldsK, kbase + (size_t)kb * 32 * HD_, 32, 32, 32, 4, 3);
        // V tile: 64 rows x 64B, global row stride 4096B; LDS rows padded to 80B
        tdm_load_2d(ldsV, vbase + kb * 32, 16, 64, 1024, 3, 3);
    }
#endif

    while (kb < NKB) {
        const int s0 = kb * 32;
        int kbn = kb + 1;
        while (kbn < NKB && !needed(kbn)) ++kbn;

#ifdef HAVE_TDM
        if (wid == 0) {
            if (kbn < NKB) {   // prefetch next stage into the other buffer
                unsigned ob = (unsigned)(buf ^ 1);
                tdm_load_2d(ldsK + ob * (KSZ * 2), kbase + (size_t)kbn * 32 * HD_,
                            32, 32, 32, 4, 3);
                tdm_load_2d(ldsV + ob * (VSZ * 2), vbase + kbn * 32,
                            16, 64, 1024, 3, 3);
                wait_tensorcnt2();   // in-order: current stage's pair has landed
            } else {
                wait_tensorcnt0();
            }
        }
#else
        // Fallback: cooperative synchronous copy of stage kb into buf
#pragma unroll
        for (int c = tid; c < 256; c += 128) {
            int row = c >> 3, col = (c & 7) * 8;
            *(bf16x8*)(Ks + buf * KSZ + row * KST + col) =
                *(const bf16x8*)(kbase + (size_t)(s0 + row) * HD_ + col);
        }
#pragma unroll
        for (int c = tid; c < 256; c += 128) {
            int row = c >> 2, col = (c & 3) * 8;
            *(bf16x8*)(Vs + buf * VSZ + row * VST + col) =
                *(const bf16x8*)(vbase + (size_t)row * L_ + s0 + col);
        }
#endif
        __syncthreads();   // publish stage buf to all 4 waves

        const __bf16* Kb = Ks + buf * KSZ;
        const __bf16* Vb = Vs + buf * VSZ;

        float sc[2][8];
#pragma unroll
        for (int ct = 0; ct < 2; ++ct) {
            const __bf16* kr = Kb + (ct * 16 + lr) * KST;
            bf16x16 bk0 = *(const bf16x16*)(kr + half * 16);       // hd 0..31
            bf16x16 bk1 = *(const bf16x16*)(kr + 32 + half * 16);  // hd 32..63
            f32x8 accs = {};
            accs = wmma_bf16(aq0, bk0, accs);
            accs = wmma_bf16(aq1, bk1, accs);

            const int  scol  = s0 + ct * 16 + lr;
            const bool sglob = (scol == gpv.x) | (scol == gpv.y) |
                               (scol == gpv.z) | (scol == gpv.w);
#pragma unroll
            for (int r = 0; r < 8; ++r) {
                int  mrow = l0 + half * 8 + r;
                int  d    = mrow - scol;
                bool ok   = (d <= WIN_ && d >= -WIN_) | rg[r] | sglob;
                sc[ct][r] = ok ? accs[r] * 0.125f : -1e30f;   // 1/sqrt(64)
            }
        }

        float alpha[8];
#pragma unroll
        for (int r = 0; r < 8; ++r) {
            float mx = fmaxf(sc[0][r], sc[1][r]);
#pragma unroll
            for (int off = 1; off < 16; off <<= 1) mx = fmaxf(mx, __shfl_xor(mx, off, 32));
            float mnew = fmaxf(mrun[r], mx);
            float a_   = __expf(mrun[r] - mnew);
            float p0   = (sc[0][r] < -1e29f) ? 0.f : __expf(sc[0][r] - mnew);
            float p1   = (sc[1][r] < -1e29f) ? 0.f : __expf(sc[1][r] - mnew);
            float ps   = p0 + p1;
#pragma unroll
            for (int off = 1; off < 16; off <<= 1) ps += __shfl_xor(ps, off, 32);
            lrun[r]  = lrun[r] * a_ + ps;
            mrun[r]  = mnew;
            alpha[r] = a_;
            Pw[(half * 8 + r) * PST + lr]      = (__bf16)p0;
            Pw[(half * 8 + r) * PST + 16 + lr] = (__bf16)p1;
        }
#pragma unroll
        for (int nt = 0; nt < 4; ++nt)
#pragma unroll
            for (int r = 0; r < 8; ++r) o[nt][r] *= alpha[r];

        // per-wave P re-layout: same-wave DS ops are processed in order
        bf16x16 ap = mk16(*(const bf16x8*)(Pw + lr * PST + half * 16),
                          *(const bf16x8*)(Pw + lr * PST + 8 + half * 16));

#pragma unroll
        for (int nt = 0; nt < 4; ++nt) {
            bf16x16 bvv = *(const bf16x16*)(Vb + (nt * 16 + lr) * VST + half * 16);
            o[nt] = wmma_bf16(ap, bvv, o[nt]);
        }

        __syncthreads();   // all reads of buf done before it is re-staged
        kb = kbn;
        buf ^= 1;
    }

#pragma unroll
    for (int r = 0; r < 8; ++r) {
        float inv  = 1.0f / lrun[r];
        int   mrow = l0 + half * 8 + r;
        float* orow = attn + ((size_t)b * L_ + mrow) * D_ + h * HD_ + lr;
#pragma unroll
        for (int nt = 0; nt < 4; ++nt) orow[nt * 16] = o[nt][r] * inv;
    }
}

// ---------------------------------------------------------------------------
// Kernel 3: output projection, 32(M) x 64(N) per wave (8 WMMA per K-chunk)
// ---------------------------------------------------------------------------
__global__ __launch_bounds__(32) void oproj_k(
    const float* __restrict__ attn, const __bf16* __restrict__ WoT,
    const float* __restrict__ bo, float* __restrict__ out)
{
    const int lane = threadIdx.x;
    const int lr   = lane & 15;
    const int half = lane >> 4;
    const int n0   = blockIdx.x * 64;
    const int m0   = blockIdx.y * 32;

    f32x8 acc[2][4] = {};
    const float*  arow0 = attn + (size_t)(m0 + lr) * D_ + half * 16;
    const float*  arow1 = attn + (size_t)(m0 + 16 + lr) * D_ + half * 16;
    const __bf16* bcol  = WoT  + (size_t)(n0 + lr) * 1024 + half * 16;

    for (int kc = 0; kc < 1024; kc += 32) {
        bf16x16 a0 = cvt16(arow0 + kc);
        bf16x16 a1 = cvt16(arow1 + kc);
#pragma unroll
        for (int nt = 0; nt < 4; ++nt) {
            bf16x16 bm = *(const bf16x16*)(bcol + (size_t)nt * 16 * 1024 + kc);
            acc[0][nt] = wmma_bf16(a0, bm, acc[0][nt]);
            acc[1][nt] = wmma_bf16(a1, bm, acc[1][nt]);
        }
    }

#pragma unroll
    for (int nt = 0; nt < 4; ++nt) {
        const float bias = bo[n0 + nt * 16 + lr];
#pragma unroll
        for (int mt2 = 0; mt2 < 2; ++mt2)
#pragma unroll
            for (int r = 0; r < 8; ++r)
                out[(size_t)(m0 + mt2 * 16 + half * 8 + r) * D_ + n0 + nt * 16 + lr] =
                    acc[mt2][nt][r] + bias;
    }
}

// ---------------------------------------------------------------------------
extern "C" void kernel_launch(void* const* d_in, const int* in_sizes, int n_in,
                              void* d_out, int out_size, void* d_ws, size_t ws_size,
                              hipStream_t stream) {
    const float* x  = (const float*)d_in[0];
    const int*   gp = (const int*)d_in[1];
    const float* Wq = (const float*)d_in[2];
    const float* bq = (const float*)d_in[3];
    const float* Wk = (const float*)d_in[4];
    const float* bk = (const float*)d_in[5];
    const float* Wv = (const float*)d_in[6];
    const float* bv = (const float*)d_in[7];
    const float* Wo = (const float*)d_in[8];
    const float* bo = (const float*)d_in[9];
    float* out = (float*)d_out;

    char* ws = (char*)d_ws;
    __bf16* WqT  = (__bf16*)(ws + WQT_OFF);
    __bf16* WkT  = (__bf16*)(ws + WKT_OFF);
    __bf16* WvT  = (__bf16*)(ws + WVT_OFF);
    __bf16* WoT  = (__bf16*)(ws + WOT_OFF);
    __bf16* qbf  = (__bf16*)(ws + QBF_OFF);
    __bf16* kbf  = (__bf16*)(ws + KBF_OFF);
    __bf16* vTb  = (__bf16*)(ws + VTB_OFF);
    float*  attn = (float*)(ws + ATTN_OFF);

    wconv_k<<<4096, 256, 0, stream>>>(Wq, WqT, 1024);
    wconv_k<<<1024, 256, 0, stream>>>(Wk, WkT, 256);
    wconv_k<<<1024, 256, 0, stream>>>(Wv, WvT, 256);
    wconv_k<<<4096, 256, 0, stream>>>(Wo, WoT, 1024);   // fixed: was Wv/WvT

    qkv_rope_k<<<dim3(24, 128), 32, 0, stream>>>(x, WqT, WkT, WvT, bq, bk, bv,
                                                 qbf, kbf, vTb);
    attn_k<<<dim3(L_ / 16, G_, B_), 128, 0, stream>>>(qbf, kbf, vTb, gp, attn);
    oproj_k<<<dim3(16, 128), 32, 0, stream>>>(attn, WoT, bo, out);
}
